// ParameterGenerationHead_36687610642946
// MI455X (gfx1250) — compile-verified, matrix operand order checked by reading, not run on hardware
//
#include <hip/hip_runtime.h>
#include <cstdint>
#include <cstddef>

typedef __bf16 bf16;
typedef __attribute__((ext_vector_type(16))) bf16 v16bf;
typedef __attribute__((ext_vector_type(8)))  bf16 v8bf;
typedef __attribute__((ext_vector_type(8)))  float v8f;
typedef __attribute__((ext_vector_type(4)))  float vf4;
typedef __attribute__((ext_vector_type(4)))  int   v4i;

#define DEV static __device__ __forceinline__

// ---- CDNA5 async global->LDS copy (ASYNCcnt path), guarded by __has_builtin ----
#if defined(__has_builtin)
#if __has_builtin(__builtin_amdgcn_global_load_async_to_lds_b128)
#define HAVE_ASYNC_COPY 1
#endif
#endif
#ifndef HAVE_ASYNC_COPY
#define HAVE_ASYNC_COPY 0
#endif

#if HAVE_ASYNC_COPY
#if __has_builtin(__builtin_amdgcn_s_wait_asynccnt)
#define WAIT_ASYNC() __builtin_amdgcn_s_wait_asynccnt(0)
#else
#define WAIT_ASYNC() asm volatile("s_wait_asynccnt 0x0" ::: "memory")
#endif
// builtin signature (from clang diagnostics): (v4i __device__*, v4i __shared__*, Ii offset, Ii cpol)
// generic LDS pointer -> addrspace(3): low 32 bits of a generic LDS address are
// the LDS byte offset (ISA 10.2)
#define AS3V4(p) ((__attribute__((address_space(3))) v4i*)(uintptr_t)(uint32_t)(uintptr_t)(p))
#define AS1V4(p) ((__attribute__((address_space(1))) v4i*)(uintptr_t)(p))
#else
#define WAIT_ASYNC()
#endif

DEV v8f vzero8() {
  v8f z;
#pragma unroll
  for (int i = 0; i < 8; ++i) z[i] = 0.f;
  return z;
}

DEV v16bf ldfrag(const bf16* p0, const bf16* p1) {
  union { v16bf v; v8bf h[2]; } u;
  u.h[0] = *(const v8bf*)p0;
  u.h[1] = *(const v8bf*)p1;
  return u.v;
}

DEV v8f wmma_bf16(v16bf a, v16bf b, v8f c) {
  return __builtin_amdgcn_wmma_f32_16x16x32_bf16(
      false, a, false, b, (short)0, c, false, false);
}

// ---------------------------------------------------------------------------
// f32 -> bf16 bulk convert (weights; once per launch). NT loads: read-once f32.
// ---------------------------------------------------------------------------
__global__ void k_cvt(const float* __restrict__ src, bf16* __restrict__ dst,
                      long long n4) {
  long long i = (long long)blockIdx.x * blockDim.x + threadIdx.x;
  long long stride = (long long)gridDim.x * blockDim.x;
  for (; i < n4; i += stride) {
    vf4 f = __builtin_nontemporal_load((const vf4*)src + i);
    bf16* d = dst + i * 4;
    d[0] = (bf16)f.x; d[1] = (bf16)f.y; d[2] = (bf16)f.z; d[3] = (bf16)f.w;
  }
}

// ---------------------------------------------------------------------------
// x[b,s,:] = token_emb[ids[b,s],:] + pos_emb[s,:]
// ---------------------------------------------------------------------------
__global__ __launch_bounds__(256) void k_embed(const int* __restrict__ ids,
                                               const float* __restrict__ tok,
                                               const float* __restrict__ pos,
                                               float* __restrict__ x,
                                               int S, int H) {
  const int row = blockIdx.x;
  const int s = row % S;
  const int id = ids[row];
  const float* tr = tok + (size_t)id * H;
  const float* pr = pos + (size_t)s * H;
  float* xr = x + (size_t)row * H;
#pragma unroll
  for (int i = 0; i < 4; ++i) {
    int c = threadIdx.x + 256 * i;
    xr[c] = tr[c] + pr[c];
  }
}

// ---------------------------------------------------------------------------
// x = LayerNorm(x + res) * g + b over H=1024; bcast: res is (B,H) over S.
// ---------------------------------------------------------------------------
__global__ __launch_bounds__(256) void k_ln(float* __restrict__ x,
                                            const float* __restrict__ res,
                                            const float* __restrict__ g,
                                            const float* __restrict__ bb,
                                            int S, int H, int bcast) {
  const int row = blockIdx.x;
  const int tid = threadIdx.x;
  const int wave = tid >> 5, lane = tid & 31;
  const size_t rrow = bcast ? (size_t)(row / S) : (size_t)row;
  float* xr = x + (size_t)row * H;
  const float* rr = res + rrow * H;

  float v[4];
  float sum = 0.f, sq = 0.f;
#pragma unroll
  for (int i = 0; i < 4; ++i) {
    int c = tid + 256 * i;
    v[i] = xr[c] + rr[c];
    sum += v[i];
    sq += v[i] * v[i];
  }
#pragma unroll
  for (int o = 16; o > 0; o >>= 1) {
    sum += __shfl_xor(sum, o, 32);
    sq  += __shfl_xor(sq,  o, 32);
  }
  __shared__ float w1[8], w2[8];
  if (lane == 0) { w1[wave] = sum; w2[wave] = sq; }
  __syncthreads();
  float ts = 0.f, tq = 0.f;
#pragma unroll
  for (int i = 0; i < 8; ++i) { ts += w1[i]; tq += w2[i]; }
  const float mean = ts / (float)H;
  const float var = tq / (float)H - mean * mean;
  const float rs = rsqrtf(var + 1e-5f);
#pragma unroll
  for (int i = 0; i < 4; ++i) {
    int c = tid + 256 * i;
    xr[c] = (v[i] - mean) * rs * g[c] + bb[c];
  }
}

// ---------------------------------------------------------------------------
// C(M,N) = A(M,K) @ W(N,K)^T + bias [+ ReLU]
// A: f32 (converted to bf16 while staging), W: bf16 (pre-converted).
// 8 waves, 128x128 block tile, K-step 32, double-buffered LDS, software
// pipeline: next-tile loads issued before current-tile WMMAs.
// W staging uses global_load_async_to_lds_b128 (ASYNCcnt) when available.
// nt: nontemporal C stores (final logits: keep L2 for weights).
// ---------------------------------------------------------------------------
__global__ __launch_bounds__(256) void k_gemm(const float* __restrict__ A,
                                              const bf16* __restrict__ W,
                                              const float* __restrict__ bias,
                                              float* __restrict__ C,
                                              int M, int N, int K,
                                              int relu, int nt) {
  constexpr int BM = 128, BN = 128, BK = 32, LDA = 40;  // 80B rows: 16B aligned, bank-safe
  __shared__ __align__(16) bf16 sA[2][BM][LDA];
  __shared__ __align__(16) bf16 sB[2][BN][LDA];

  const int tid = threadIdx.x;
  const int wave = tid >> 5, lane = tid & 31;
  const int half = lane >> 4, idx = lane & 15;
  const int bm = blockIdx.y * BM, bn = blockIdx.x * BN;
  const int wm = (wave >> 1) * 32, wn = (wave & 1) * 64;

  const int sr = tid >> 1;        // staging row 0..127
  const int sc = (tid & 1) * 16;  // staging col 0 | 16
  const bool okA = (bm + sr) < M;
  const bool okB = (bn + sr) < N;
  const float* gA = A + (size_t)(bm + sr) * K + sc;
  const bf16*  gW = W + (size_t)(bn + sr) * K + sc;

  // ---- prologue: zero-fill OOB rows once (both buffers), stage tile 0 ----
  if (!okA) {
#pragma unroll
    for (int j = 0; j < 16; ++j) { sA[0][sr][sc + j] = (bf16)0.f; sA[1][sr][sc + j] = (bf16)0.f; }
  } else {
    bf16* dst = &sA[0][sr][sc];
#pragma unroll
    for (int j = 0; j < 4; ++j) {
      float4 f = *((const float4*)gA + j);
      dst[j * 4 + 0] = (bf16)f.x; dst[j * 4 + 1] = (bf16)f.y;
      dst[j * 4 + 2] = (bf16)f.z; dst[j * 4 + 3] = (bf16)f.w;
    }
  }
  if (!okB) {
    v8bf z;
#pragma unroll
    for (int j = 0; j < 8; ++j) z[j] = (bf16)0.f;
    ((v8bf*)&sB[0][sr][sc])[0] = z; ((v8bf*)&sB[0][sr][sc])[1] = z;
    ((v8bf*)&sB[1][sr][sc])[0] = z; ((v8bf*)&sB[1][sr][sc])[1] = z;
  } else {
#if HAVE_ASYNC_COPY
    __builtin_amdgcn_global_load_async_to_lds_b128(AS1V4(gW), AS3V4(&sB[0][sr][sc]), 0, 0);
    __builtin_amdgcn_global_load_async_to_lds_b128(AS1V4(gW + 8), AS3V4(&sB[0][sr][sc + 8]), 0, 0);
#else
    ((v8bf*)&sB[0][sr][sc])[0] = ((const v8bf*)gW)[0];
    ((v8bf*)&sB[0][sr][sc])[1] = ((const v8bf*)gW)[1];
#endif
  }
  WAIT_ASYNC();
  __syncthreads();

  v8f acc[2][4];
#pragma unroll
  for (int i = 0; i < 2; ++i)
#pragma unroll
    for (int j = 0; j < 4; ++j) acc[i][j] = vzero8();

  for (int kt = 0; kt < K; kt += BK) {
    const int cur = (kt / BK) & 1, nxt = cur ^ 1;
    const bool more = (kt + BK) < K;

    // ---- issue next-tile loads before computing on current tile ----
    float4 fa[4];
    if (more) {
      if (okA) {
#pragma unroll
        for (int j = 0; j < 4; ++j) fa[j] = *((const float4*)(gA + kt + BK) + j);
      }
#if HAVE_ASYNC_COPY
      if (okB) {
        const bf16* ws = gW + kt + BK;
        __builtin_amdgcn_global_load_async_to_lds_b128(AS1V4(ws), AS3V4(&sB[nxt][sr][sc]), 0, 0);
        __builtin_amdgcn_global_load_async_to_lds_b128(AS1V4(ws + 8), AS3V4(&sB[nxt][sr][sc + 8]), 0, 0);
      }
#endif
    }
#if !HAVE_ASYNC_COPY
    v8bf wb0, wb1;
    if (more && okB) {
      wb0 = ((const v8bf*)(gW + kt + BK))[0];
      wb1 = ((const v8bf*)(gW + kt + BK))[1];
    }
#endif

    // ---- WMMA on current tile ----
    v16bf af[2], bfr[4];
#pragma unroll
    for (int i = 0; i < 2; ++i) {
      const bf16* r0 = &sA[cur][wm + i * 16 + idx][half * 8];
      af[i] = ldfrag(r0, r0 + 16);
    }
#pragma unroll
    for (int j = 0; j < 4; ++j) {
      const bf16* r0 = &sB[cur][wn + j * 16 + idx][half * 8];
      bfr[j] = ldfrag(r0, r0 + 16);
    }
#pragma unroll
    for (int i = 0; i < 2; ++i)
#pragma unroll
      for (int j = 0; j < 4; ++j) acc[i][j] = wmma_bf16(af[i], bfr[j], acc[i][j]);

    // ---- commit staged next tile ----
    if (more) {
      if (okA) {
        bf16* dst = &sA[nxt][sr][sc];
#pragma unroll
        for (int j = 0; j < 4; ++j) {
          dst[j * 4 + 0] = (bf16)fa[j].x; dst[j * 4 + 1] = (bf16)fa[j].y;
          dst[j * 4 + 2] = (bf16)fa[j].z; dst[j * 4 + 3] = (bf16)fa[j].w;
        }
      }
#if !HAVE_ASYNC_COPY
      if (okB) {
        ((v8bf*)&sB[nxt][sr][sc])[0] = wb0;
        ((v8bf*)&sB[nxt][sr][sc])[1] = wb1;
      }
#endif
    }
    WAIT_ASYNC();
    __syncthreads();
  }

  // ---- epilogue: bias (+ReLU), optional nontemporal stores ----
#pragma unroll
  for (int i = 0; i < 2; ++i)
#pragma unroll
    for (int j = 0; j < 4; ++j) {
      const int colg = bn + wn + j * 16 + idx;
      if (colg < N) {
        const float bv = bias[colg];
#pragma unroll
        for (int r = 0; r < 8; ++r) {
          const int rowg = bm + wm + i * 16 + half * 8 + r;
          if (rowg < M) {
            float val = acc[i][j][r] + bv;
            if (relu) val = fmaxf(val, 0.f);
            float* cp = C + (size_t)rowg * N + colg;
            if (nt) __builtin_nontemporal_store(val, cp);
            else *cp = val;
          }
        }
      }
    }
}

// ---------------------------------------------------------------------------
// Causal self-attention for one (b,h): S=128, HD=64.
// ---------------------------------------------------------------------------
__global__ __launch_bounds__(256) void k_attn(const float* __restrict__ qkv,
                                              float* __restrict__ o,
                                              int NH) {
  constexpr int S = 128, HD = 64, H = 1024, H3 = 3 * H;
  constexpr float SCALE = 0.125f;  // 1/sqrt(64)
  constexpr int LQ = HD + 8;       // 144B rows
  constexpr int LV = S + 8;        // 272B rows
  constexpr int LS = S + 1;        // conflict-free f32 columns

  __shared__ __align__(16) bf16 sQ[S][LQ];
  __shared__ __align__(16) bf16 sK[S][LQ];
  __shared__ __align__(16) bf16 sVT[HD][LV];
  __shared__ __align__(16) bf16 sAt[S][LV];
  __shared__ float sS[S][LS];

  const int b = blockIdx.x / NH, h = blockIdx.x % NH;
  const int tid = threadIdx.x;
  const int wave = tid >> 5, lane = tid & 31;
  const int half = lane >> 4, idx = lane & 15;
  const int wm = wave * 16;

  // ---- stage q (scaled), k, v^T ----
  {
    const int r = tid >> 1, c0 = (tid & 1) * 32;
    const float* base = qkv + (size_t)(b * S + r) * H3 + h * HD + c0;
#pragma unroll
    for (int j = 0; j < 8; ++j) {
      float4 f = *(const float4*)(base + j * 4);
      sQ[r][c0 + j * 4 + 0] = (bf16)(f.x * SCALE);
      sQ[r][c0 + j * 4 + 1] = (bf16)(f.y * SCALE);
      sQ[r][c0 + j * 4 + 2] = (bf16)(f.z * SCALE);
      sQ[r][c0 + j * 4 + 3] = (bf16)(f.w * SCALE);
    }
#pragma unroll
    for (int j = 0; j < 8; ++j) {
      float4 f = *(const float4*)(base + H + j * 4);
      sK[r][c0 + j * 4 + 0] = (bf16)f.x;
      sK[r][c0 + j * 4 + 1] = (bf16)f.y;
      sK[r][c0 + j * 4 + 2] = (bf16)f.z;
      sK[r][c0 + j * 4 + 3] = (bf16)f.w;
    }
#pragma unroll
    for (int j = 0; j < 8; ++j) {
      float4 f = *(const float4*)(base + 2 * H + j * 4);
      sVT[c0 + j * 4 + 0][r] = (bf16)f.x;
      sVT[c0 + j * 4 + 1][r] = (bf16)f.y;
      sVT[c0 + j * 4 + 2][r] = (bf16)f.z;
      sVT[c0 + j * 4 + 3][r] = (bf16)f.w;
    }
  }
  __syncthreads();

  // ---- phase 1: scores = (q*scale) @ k^T, causal mask at writeback ----
  {
    v8f acc[8];
#pragma unroll
    for (int ntt = 0; ntt < 8; ++ntt) acc[ntt] = vzero8();
#pragma unroll
    for (int kt = 0; kt < HD; kt += 32) {
      const bf16* qa = &sQ[wm + idx][kt + half * 8];
      v16bf aq = ldfrag(qa, qa + 16);
#pragma unroll
      for (int ntt = 0; ntt < 8; ++ntt) {
        const bf16* kb = &sK[ntt * 16 + idx][kt + half * 8];
        acc[ntt] = wmma_bf16(aq, ldfrag(kb, kb + 16), acc[ntt]);
      }
    }
#pragma unroll
    for (int ntt = 0; ntt < 8; ++ntt)
#pragma unroll
      for (int r = 0; r < 8; ++r) {
        const int row = wm + half * 8 + r;
        const int col = ntt * 16 + idx;
        sS[row][col] = (col <= row) ? acc[ntt][r] : -1e30f;
      }
  }
  __syncthreads();

  // ---- phase 2: row softmax (16 rows per wave, shfl reductions) ----
  for (int r = 0; r < 16; ++r) {
    const int m = wm + r;
    float v0 = sS[m][lane], v1 = sS[m][lane + 32];
    float v2 = sS[m][lane + 64], v3 = sS[m][lane + 96];
    float mx = fmaxf(fmaxf(v0, v1), fmaxf(v2, v3));
#pragma unroll
    for (int off = 16; off > 0; off >>= 1) mx = fmaxf(mx, __shfl_xor(mx, off, 32));
    v0 = __expf(v0 - mx); v1 = __expf(v1 - mx);
    v2 = __expf(v2 - mx); v3 = __expf(v3 - mx);
    float sum = v0 + v1 + v2 + v3;
#pragma unroll
    for (int off = 16; off > 0; off >>= 1) sum += __shfl_xor(sum, off, 32);
    const float inv = 1.f / sum;
    sAt[m][lane]      = (bf16)(v0 * inv);
    sAt[m][lane + 32] = (bf16)(v1 * inv);
    sAt[m][lane + 64] = (bf16)(v2 * inv);
    sAt[m][lane + 96] = (bf16)(v3 * inv);
  }
  __syncthreads();

  // ---- phase 3: O = attn @ v ----
  {
    v8f oacc[4];
#pragma unroll
    for (int ntt = 0; ntt < 4; ++ntt) oacc[ntt] = vzero8();
#pragma unroll
    for (int kt = 0; kt < S; kt += 32) {
      const bf16* aa = &sAt[wm + idx][kt + half * 8];
      v16bf af = ldfrag(aa, aa + 16);
#pragma unroll
      for (int ntt = 0; ntt < 4; ++ntt) {
        const bf16* vb = &sVT[ntt * 16 + idx][kt + half * 8];
        oacc[ntt] = wmma_bf16(af, ldfrag(vb, vb + 16), oacc[ntt]);
      }
    }
#pragma unroll
    for (int ntt = 0; ntt < 4; ++ntt)
#pragma unroll
      for (int r = 0; r < 8; ++r) {
        const int row = wm + half * 8 + r;
        const int col = ntt * 16 + idx;
        o[(size_t)(b * S + row) * H + h * HD + col] = oacc[ntt][r];
      }
  }
}

// ---------------------------------------------------------------------------
// Host orchestration
// ---------------------------------------------------------------------------
extern "C" void kernel_launch(void* const* d_in, const int* in_sizes, int n_in,
                              void* d_out, int out_size, void* d_ws, size_t ws_size,
                              hipStream_t stream) {
  constexpr int B = 16, S = 128, AD = 512, H = 1024, NH = 16, L = 4, V = 32000;
  constexpr int M = B * S;
  constexpr int H3 = 3 * H, H4 = 4 * H;

  const float* y_state    = (const float*)d_in[0];
  const int*   target_ids = (const int*)d_in[1];
  const float* token_emb  = (const float*)d_in[2];
  const float* pos_emb    = (const float*)d_in[3];
  const float* ctx_W      = (const float*)d_in[4];
  const float* ctx_b      = (const float*)d_in[5];
  const float* Wqkv_self  = (const float*)d_in[6];
  const float* bqkv_self  = (const float*)d_in[7];
  const float* Wo_self    = (const float*)d_in[8];
  const float* bo_self    = (const float*)d_in[9];
  const float* Wqkv_cross = (const float*)d_in[10];
  const float* bqkv_cross = (const float*)d_in[11];
  const float* Wo_cross   = (const float*)d_in[12];
  const float* bo_cross   = (const float*)d_in[13];
  const float* W1         = (const float*)d_in[14];
  const float* b1         = (const float*)d_in[15];
  const float* W2         = (const float*)d_in[16];
  const float* b2         = (const float*)d_in[17];
  const float* ln1_g      = (const float*)d_in[18];
  const float* ln1_b      = (const float*)d_in[19];
  const float* ln2_g      = (const float*)d_in[20];
  const float* ln2_b      = (const float*)d_in[21];
  const float* ln3_g      = (const float*)d_in[22];
  const float* ln3_b      = (const float*)d_in[23];
  const float* out_W      = (const float*)d_in[24];
  const float* out_b      = (const float*)d_in[25];
  float* out = (float*)d_out;

  char* wsb = (char*)d_ws;
  size_t off = 0;
  auto alloc = [&](size_t bytes) -> void* {
    void* p = wsb + off;
    off += (bytes + 255) & ~(size_t)255;
    return p;
  };

  float* x   = (float*)alloc((size_t)M * H * 4);
  float* qkv = (float*)alloc((size_t)M * H3 * 4);
  float* att = (float*)alloc((size_t)M * H * 4);
  float* tmp = (float*)alloc((size_t)M * H * 4);
  float* h1  = (float*)alloc((size_t)M * H4 * 4);
  float* mem = (float*)alloc((size_t)B * H * 4);
  float* cv  = (float*)alloc((size_t)B * H * 4);
  float* cc  = (float*)alloc((size_t)B * H * 4);

  bf16* ctxW_b = (bf16*)alloc((size_t)H * AD * 2);
  bf16* qkvs_b = (bf16*)alloc((size_t)L * H3 * H * 2);
  bf16* wos_b  = (bf16*)alloc((size_t)L * H * H * 2);
  bf16* wvc_b  = (bf16*)alloc((size_t)L * H * H * 2);
  bf16* woc_b  = (bf16*)alloc((size_t)L * H * H * 2);
  bf16* w1_b   = (bf16*)alloc((size_t)L * H4 * H * 2);
  bf16* w2_b   = (bf16*)alloc((size_t)L * H * H4 * 2);
  bf16* outW_b = (bf16*)alloc((size_t)V * H * 2);

  auto cvt = [&](const float* s, bf16* d, size_t n) {
    long long n4 = (long long)(n / 4);
    int grid = (int)((n4 + 255) / 256);
    if (grid > 16384) grid = 16384;
    k_cvt<<<grid, 256, 0, stream>>>(s, d, n4);
  };
  auto gemm = [&](const float* Aa, const bf16* Ww, const float* bias, float* Cc,
                  int Mm, int Nn, int Kk, int relu, int nt) {
    dim3 g((Nn + 127) / 128, (Mm + 127) / 128);
    k_gemm<<<g, 256, 0, stream>>>(Aa, Ww, bias, Cc, Mm, Nn, Kk, relu, nt);
  };

  // one-time weight conversion -> bf16 (L2-resident thereafter)
  cvt(ctx_W, ctxW_b, (size_t)H * AD);
  cvt(Wqkv_self, qkvs_b, (size_t)L * H3 * H);
  cvt(Wo_self, wos_b, (size_t)L * H * H);
  for (int l = 0; l < L; ++l)  // cross-attn: only Wv matters (softmax over Sk=1 == 1)
    cvt(Wqkv_cross + (size_t)l * H3 * H + (size_t)2 * H * H,
        wvc_b + (size_t)l * H * H, (size_t)H * H);
  cvt(Wo_cross, woc_b, (size_t)L * H * H);
  cvt(W1, w1_b, (size_t)L * H4 * H);
  cvt(W2, w2_b, (size_t)L * H * H4);
  cvt(out_W, outW_b, (size_t)V * H);

  // embeddings and context memory
  k_embed<<<M, 256, 0, stream>>>(target_ids, token_emb, pos_emb, x, S, H);
  gemm(y_state, ctxW_b, ctx_b, mem, B, H, AD, 0, 0);

  for (int l = 0; l < L; ++l) {
    // self-attention
    gemm(x, qkvs_b + (size_t)l * H3 * H, bqkv_self + (size_t)l * H3, qkv, M, H3, H, 0, 0);
    k_attn<<<B * NH, 256, 0, stream>>>(qkv, att, NH);
    gemm(att, wos_b + (size_t)l * H * H, bo_self + (size_t)l * H, tmp, M, H, H, 0, 0);
    k_ln<<<M, 256, 0, stream>>>(x, tmp, ln1_g + (size_t)l * H, ln1_b + (size_t)l * H, S, H, 0);

    // cross-attention (Sk=1): out = (mem @ Wv^T + bv) @ Wo^T + bo, broadcast over S
    gemm(mem, wvc_b + (size_t)l * H * H, bqkv_cross + (size_t)l * H3 + 2 * H, cv, B, H, H, 0, 0);
    gemm(cv, woc_b + (size_t)l * H * H, bo_cross + (size_t)l * H, cc, B, H, H, 0, 0);
    k_ln<<<M, 256, 0, stream>>>(x, cc, ln2_g + (size_t)l * H, ln2_b + (size_t)l * H, S, H, 1);

    // FFN
    gemm(x, w1_b + (size_t)l * H4 * H, b1 + (size_t)l * H4, h1, M, H4, H, 1, 0);
    gemm(h1, w2_b + (size_t)l * H * H4, b2 + (size_t)l * H, tmp, M, H, H4, 0, 0);
    k_ln<<<M, 256, 0, stream>>>(x, tmp, ln3_g + (size_t)l * H, ln3_b + (size_t)l * H, S, H, 0);
  }

  // logits: stream 262MB out with NT stores (don't evict L2-resident weights)
  gemm(x, outW_b, out_b, out, M, V, H, 0, 1);
}
